// GNN_25829933318528
// MI455X (gfx1250) — compile-verified, matrix operand order checked by reading, not run on hardware
//
#include <hip/hip_runtime.h>
#include <hip/hip_bf16.h>

// ---------------------------------------------------------------------------
// GNN forward on MI455X (gfx1250): bf16 WMMA GEMMs (LDS-staged, software-
// pipelined, LDS-only split barriers) + L2-atomic SpMM.
// ---------------------------------------------------------------------------

typedef __bf16  v16bf __attribute__((ext_vector_type(16)));
typedef float   v8f   __attribute__((ext_vector_type(8)));

static constexpr int    N_NODES = 50000;
static constexpr int    N_EDGES = 800000;
static constexpr int    NPAD    = 50048;          // 391 * 128 (M-tile multiple)
static constexpr int    IN_DIM  = 512, H1D = 512, H2D = 256, OUT_DIM = 128;

// Workspace layout (bytes)
static constexpr size_t OFF_XBF  = 0;                                    // [NPAD][512] bf16
static constexpr size_t OFF_H1   = OFF_XBF  + (size_t)NPAD * 512 * 2;    // [NPAD][512] bf16
static constexpr size_t OFF_T    = OFF_H1   + (size_t)NPAD * 512 * 2;    // [NPAD][256] f32
static constexpr size_t OFF_S    = OFF_T    + (size_t)NPAD * 256 * 4;    // [NPAD][256] f32
static constexpr size_t OFF_H2   = OFF_S    + (size_t)NPAD * 256 * 4;    // [NPAD][256] bf16
static constexpr size_t OFF_W1T  = OFF_H2   + (size_t)NPAD * 256 * 2;    // [512][512] bf16
static constexpr size_t OFF_WG1T = OFF_W1T  + (size_t)512 * 512 * 2;     // [256][512] bf16
static constexpr size_t OFF_WG2T = OFF_WG1T + (size_t)256 * 512 * 2;     // [256][256] bf16
static constexpr size_t OFF_WL2T = OFF_WG2T + (size_t)256 * 256 * 2;     // [128][256] bf16

union FragBF { v16bf v; uint4 q[2]; };

// LDS-only workgroup barrier: drains this wave's DS ops, then split
// signal/wait on the workgroup barrier. Unlike __syncthreads(), it does NOT
// drain LOADcnt, so register-destined global prefetches stay in flight
// across the barrier (their s_wait_loadcnt lands at first use).
__device__ __forceinline__ void lds_barrier()
{
    asm volatile("s_wait_dscnt 0x0\n\t"
                 "s_barrier_signal -1\n\t"
                 "s_barrier_wait -1" ::: "memory");
}

// --------------------------- elementwise helpers ---------------------------

__global__ __launch_bounds__(256) void cvt_pad_bf16(
    const float* __restrict__ X, __bf16* __restrict__ Xb,
    int rows, int cols, int rowspad)
{
    size_t i = (size_t)blockIdx.x * 256 + threadIdx.x;
    size_t tot = (size_t)rowspad * cols;
    if (i >= tot) return;
    int row = (int)(i / (size_t)cols);
    Xb[i] = (row < rows) ? (__bf16)X[i] : (__bf16)0.0f;
}

// W [K][N] f32 row-major  ->  Wt [N][K] bf16 row-major
__global__ __launch_bounds__(256) void transpose_to_bf16(
    const float* __restrict__ W, __bf16* __restrict__ Wt, int K, int N)
{
    int i = blockIdx.x * 256 + threadIdx.x;
    if (i >= K * N) return;
    int n = i / K, k = i - n * K;
    Wt[i] = (__bf16)W[(size_t)k * N + n];
}

__global__ __launch_bounds__(256) void zero_f32(float* __restrict__ p, size_t n)
{
    size_t i = (size_t)blockIdx.x * 256 + threadIdx.x;
    if (i < n) p[i] = 0.0f;
}

// H = bf16( relu(S + bias[col]) ), cols is a power of two (256)
__global__ __launch_bounds__(256) void bias_relu_to_bf16(
    const float* __restrict__ S, const float* __restrict__ bias,
    __bf16* __restrict__ H, int cols, size_t n)
{
    size_t i = (size_t)blockIdx.x * 256 + threadIdx.x;
    if (i >= n) return;
    float v = S[i] + bias[i & (size_t)(cols - 1)];
    H[i] = (__bf16)fmaxf(v, 0.0f);
}

// ------------------------------- WMMA GEMM ---------------------------------
// C[M][N] = act( A[M][K] @ B[K][N] + bias ),  A bf16 row-major, Bt[N][K] bf16.
// Block: 256 threads = 8 waves. Block tile 128x64; wave tile 16x64 (4 WMMAs).
//
// B tile (32K x 64N = 4KB) is staged in LDS in *fragment order*: staging
// thread t supplies the 16B chunk (j = t>>6, wlane = (t&63)>>1, half = t&1)
// so that consumer lane l of fragment j reads 32 contiguous bytes at
// lds[j*64 + l*2]. Double-buffered; one LDS-only barrier per k-step.
// Pipeline per step:
//   prefetch(step+1 globals)  -> barrier -> ds_load B frags -> 4x WMMA
//   -> ds_store next B chunk (max window for the prefetched load to land)
template<int K, bool BIAS, int ACT /*0 none, 1 sigmoid*/, typename OutT>
__global__ __launch_bounds__(256) void wmma_gemm_kernel(
    const __bf16* __restrict__ A,   // [>=grid.y*128][K]
    const __bf16* __restrict__ Bt,  // [N][K]
    const float*  __restrict__ bias,// [N] (unused if !BIAS)
    OutT* __restrict__ C,           // [mstore][N]
    int N, int mstore)
{
    __shared__ uint4 bstage[2][256];     // 2 x 4KB

    const int tid     = threadIdx.x;
    const int lane    = tid & 31;
    const int wave    = tid >> 5;
    const int halfsel = lane >> 4;        // 0 or 1
    const int lrow    = lane & 15;
    const int m0      = blockIdx.y * 128 + wave * 16;
    const int n0      = blockIdx.x * 64;

    // Per-thread B staging source (fragment-order chunk for k-step 0).
    const int cj = tid >> 6;              // fragment j   (0..3)
    const int cl = (tid & 63) >> 1;       // consumer lane (0..31)
    const int cc = tid & 1;               // 16B half     (0..1)
    const __bf16* bsrc = Bt + (size_t)(n0 + cj * 16 + (cl & 15)) * K
                            + ((cl >> 4) * 8 + cc * 16);

    // Per-lane A fragment source (CDNA5 16-bit A 16x32 layout).
    const __bf16* aptr = A + (size_t)(m0 + lrow) * K + halfsel * 8;

    v8f acc[4];
    const v8f vzero = {0.f,0.f,0.f,0.f,0.f,0.f,0.f,0.f};
#pragma unroll
    for (int j = 0; j < 4; ++j) acc[j] = vzero;

    constexpr int S = K / 32;

    // Prologue: fetch step-0 data; stage B chunk into buffer 0.
    uint4 breg = *(const uint4*)(bsrc);
    bstage[0][tid] = breg;
    FragBF a_cur;
    a_cur.q[0] = *(const uint4*)(aptr);
    a_cur.q[1] = *(const uint4*)(aptr + 16);

#pragma unroll
    for (int step = 0; step < S; ++step) {
        const int kk = step * 32;

        // Prefetch step+1 global data (stays in flight across the barrier).
        FragBF a_nxt;
        uint4  bnxt;
        if (step + 1 < S) {
            bnxt       = *(const uint4*)(bsrc + kk + 32);
            a_nxt.q[0] = *(const uint4*)(aptr + kk + 32);
            a_nxt.q[1] = *(const uint4*)(aptr + kk + 48);
        }

        lds_barrier();   // bstage[step&1] writes visible; prior reads drained

        // Read this step's 4 B fragments from LDS (ds_load_b128 pairs).
        const uint4* rb = &bstage[step & 1][0];
        FragBF b[4];
#pragma unroll
        for (int j = 0; j < 4; ++j) {
            b[j].q[0] = rb[j * 64 + lane * 2];
            b[j].q[1] = rb[j * 64 + lane * 2 + 1];
        }

#pragma unroll
        for (int j = 0; j < 4; ++j) {
            acc[j] = __builtin_amdgcn_wmma_f32_16x16x32_bf16(
                false, a_cur.v, false, b[j].v, (short)0, acc[j], false, false);
        }

        // Stage step+1's B chunk into the other buffer (last: max latency
        // window for the prefetched bnxt before its wait).
        if (step + 1 < S) {
            bstage[(step + 1) & 1][tid] = bnxt;
            a_cur = a_nxt;
        }
    }

    // C/D layout: VGPR r, lanes 0-15 -> (M=r, N=lane); lanes 16-31 -> (M=8+r).
    const int col0 = n0 + lrow;
#pragma unroll
    for (int j = 0; j < 4; ++j) {
        const int col = col0 + j * 16;
        const float bv = BIAS ? bias[col] : 0.0f;
#pragma unroll
        for (int r = 0; r < 8; ++r) {
            const int row = m0 + r + 8 * halfsel;
            float v = acc[j][r] + bv;
            if (ACT == 1) v = 1.0f / (1.0f + __expf(-v));
            if (row < mstore) C[(size_t)row * N + col] = (OutT)v;
        }
    }
}

// --------------------------------- SpMM ------------------------------------
// S[row[e]] += w[e] * T[col[e]]  over 256-dim rows. One wave per edge:
// 32 lanes x float4 = 128 dims per iteration, 2 iterations.
__global__ __launch_bounds__(256) void spmm_kernel(
    const int* __restrict__ erow, const int* __restrict__ ecol,
    const float* __restrict__ ew, const float* __restrict__ T,
    float* __restrict__ S, int E)
{
    const int wave = threadIdx.x >> 5;
    const int lane = threadIdx.x & 31;
    const int e = blockIdx.x * 8 + wave;
    if (e >= E) return;
    const int   r = erow[e];
    const int   c = ecol[e];
    const float w = ew[e];
    const float4* src = (const float4*)(T + (size_t)c * 256) + lane;
    float*        dst = S + (size_t)r * 256 + lane * 4;
#pragma unroll
    for (int it = 0; it < 2; ++it) {
        float4 v = src[it * 32];
        float* d = dst + it * 128;
        atomicAdd(d + 0, w * v.x);
        atomicAdd(d + 1, w * v.y);
        atomicAdd(d + 2, w * v.z);
        atomicAdd(d + 3, w * v.w);
    }
}

// ------------------------------- launcher ----------------------------------

extern "C" void kernel_launch(void* const* d_in, const int* in_sizes, int n_in,
                              void* d_out, int out_size, void* d_ws, size_t ws_size,
                              hipStream_t stream)
{
    const float* X    = (const float*)d_in[0];
    const int*   erow = (const int*)  d_in[1];
    const int*   ecol = (const int*)  d_in[2];
    const float* ew   = (const float*)d_in[3];
    const float* W1   = (const float*)d_in[4];
    const float* b1   = (const float*)d_in[5];
    const float* Wg1  = (const float*)d_in[6];
    const float* bg1  = (const float*)d_in[7];
    const float* Wg2  = (const float*)d_in[8];
    const float* bg2  = (const float*)d_in[9];
    const float* W2   = (const float*)d_in[10];
    const float* b2   = (const float*)d_in[11];
    float* out = (float*)d_out;

    char* ws = (char*)d_ws;
    __bf16* Xbf  = (__bf16*)(ws + OFF_XBF);
    __bf16* H1b  = (__bf16*)(ws + OFF_H1);
    float*  Tf   = (float*) (ws + OFF_T);
    float*  Sf   = (float*) (ws + OFF_S);
    __bf16* H2b  = (__bf16*)(ws + OFF_H2);
    __bf16* W1t  = (__bf16*)(ws + OFF_W1T);
    __bf16* Wg1t = (__bf16*)(ws + OFF_WG1T);
    __bf16* Wg2t = (__bf16*)(ws + OFF_WG2T);
    __bf16* Wl2t = (__bf16*)(ws + OFF_WL2T);

    // 1) Convert inputs once: features -> padded bf16, weights -> transposed bf16.
    {
        size_t tot = (size_t)NPAD * IN_DIM;
        cvt_pad_bf16<<<(unsigned)((tot + 255) / 256), 256, 0, stream>>>(
            X, Xbf, N_NODES, IN_DIM, NPAD);
    }
    transpose_to_bf16<<<(512 * 512 + 255) / 256, 256, 0, stream>>>(W1,  W1t,  512, 512);
    transpose_to_bf16<<<(512 * 256 + 255) / 256, 256, 0, stream>>>(Wg1, Wg1t, 512, 256);
    transpose_to_bf16<<<(256 * 256 + 255) / 256, 256, 0, stream>>>(Wg2, Wg2t, 256, 256);
    transpose_to_bf16<<<(256 * 128 + 255) / 256, 256, 0, stream>>>(W2,  Wl2t, 256, 128);

    const unsigned MB = NPAD / 128;                 // 391 row blocks
    const size_t   nD = (size_t)NPAD * H2D;         // elements in [NPAD][256]

    // 2) H1 = sigmoid(X @ W1 + b1)            [NPAD][512] bf16
    wmma_gemm_kernel<512, true, 1, __bf16>
        <<<dim3(H1D / 64, MB), 256, 0, stream>>>(Xbf, W1t, b1, H1b, H1D, NPAD);

    // 3) T = H1 @ Wg1                         [NPAD][256] f32
    wmma_gemm_kernel<512, false, 0, float>
        <<<dim3(H2D / 64, MB), 256, 0, stream>>>(H1b, Wg1t, nullptr, Tf, H2D, NPAD);

    // 4) S = spmm(T);  H2 = relu(S + bg1)     bf16
    zero_f32<<<(unsigned)((nD + 255) / 256), 256, 0, stream>>>(Sf, nD);
    spmm_kernel<<<(N_EDGES + 7) / 8, 256, 0, stream>>>(erow, ecol, ew, Tf, Sf, N_EDGES);
    bias_relu_to_bf16<<<(unsigned)((nD + 255) / 256), 256, 0, stream>>>(
        Sf, bg1, H2b, H2D, nD);

    // 5) T = H2 @ Wg2                         [NPAD][256] f32 (reuse T)
    wmma_gemm_kernel<256, false, 0, float>
        <<<dim3(H2D / 64, MB), 256, 0, stream>>>(H2b, Wg2t, nullptr, Tf, H2D, NPAD);

    // 6) S = spmm(T);  H3 = relu(S + bg2)     bf16 (reuse H2 buffer)
    zero_f32<<<(unsigned)((nD + 255) / 256), 256, 0, stream>>>(Sf, nD);
    spmm_kernel<<<(N_EDGES + 7) / 8, 256, 0, stream>>>(erow, ecol, ew, Tf, Sf, N_EDGES);
    bias_relu_to_bf16<<<(unsigned)((nD + 255) / 256), 256, 0, stream>>>(
        Sf, bg2, H2b, H2D, nD);

    // 7) out = H3 @ Wlin2 + b2                [50000][128] f32, guarded stores
    wmma_gemm_kernel<256, true, 0, float>
        <<<dim3(OUT_DIM / 64, MB), 256, 0, stream>>>(H2b, Wl2t, b2, out, OUT_DIM, N_NODES);
}
